// dense_decompo_nbcp_baens_61942018343206
// MI455X (gfx1250) — matrix operand order repaired and checked by reading, not compiled, not run on hardware
//
#include <hip/hip_runtime.h>
#include <hip/hip_bf16.h>

typedef _Float16 h16;
typedef __attribute__((ext_vector_type(4)))  float    v4f;
typedef __attribute__((ext_vector_type(8)))  float    v8f;
typedef __attribute__((ext_vector_type(8)))  _Float16 v8h;
typedef __attribute__((ext_vector_type(16))) _Float16 v16h;

#define N_ENS 8
#define D1C   1024
#define D2C   1024
#define DD    (D1C * D2C)
#define BWC   4
#define KCP   4

#if defined(__AMDGCN__) && \
    __has_builtin(__builtin_amdgcn_global_load_async_to_lds_b128) && \
    __has_builtin(__builtin_amdgcn_s_wait_asynccnt)
#define USE_ASYNC_B 1
#define AS1 __attribute__((address_space(1)))
#define AS3 __attribute__((address_space(3)))
typedef int v4i __attribute__((vector_size(16)));   // builtin's pointee type
#else
#define USE_ASYNC_B 0
#endif

// ---------------------------------------------------------------------------
// Kernel 1: decode CP-quantized soft-binary weights
//   Theta[n,b,d] = sum_k U[n,d,k] * V[b,d,k]
//   integer[n,d] = sum_b sigmoid(clip(Theta, -10, 10)) * 2^b
//   w[n,d]       = scale[n] * (integer - biasq[n] - 16)
// Written TRANSPOSED as f16: wT[n][d2][d1]  (k-contiguous for the GEMM's B op)
// ---------------------------------------------------------------------------
__global__ __launch_bounds__(256) void decode_w_kernel(
    const float* __restrict__ U, const float* __restrict__ V,
    const float* __restrict__ scale, const float* __restrict__ biasq,
    h16* __restrict__ wT)
{
  __shared__ h16 wtile[N_ENS][32][65];   // [n][d1_local][d2_local], padded
  const int t  = threadIdx.x;
  const int j0 = blockIdx.x * 64;        // d2 tile base
  const int i0 = blockIdx.y * 32;        // d1 tile base

  float sc[N_ENS], bq[N_ENS];
#pragma unroll
  for (int nn = 0; nn < N_ENS; ++nn) { sc[nn] = scale[nn]; bq[nn] = biasq[nn]; }

  const int d2l = t & 63;
  const int d1b = t >> 6;                // 0..3
#pragma unroll
  for (int r = 0; r < 8; ++r) {
    const int d1l = d1b + r * 4;
    const int d   = (i0 + d1l) * D2C + (j0 + d2l);
    v4f vb[BWC];
#pragma unroll
    for (int b = 0; b < BWC; ++b)
      vb[b] = *(const v4f*)(V + ((size_t)b * DD + d) * KCP);
#pragma unroll
    for (int nn = 0; nn < N_ENS; ++nn) {
      v4f u = *(const v4f*)(U + ((size_t)nn * DD + d) * KCP);
      float integer = 0.f;
#pragma unroll
      for (int b = 0; b < BWC; ++b) {
        float th = u[0]*vb[b][0] + u[1]*vb[b][1] + u[2]*vb[b][2] + u[3]*vb[b][3];
        th = fminf(10.f, fmaxf(-10.f, th));
        float s = 1.f / (1.f + __expf(-th));
        integer += s * (float)(1 << b);
      }
      wtile[nn][d1l][d2l] = (h16)(sc[nn] * (integer - bq[nn] - 16.f));
    }
  }
  __syncthreads();

  // transposed, coalesced write: 2 halves (adjacent d1) packed per u32
  const int p2  = (t & 15) * 2;          // d1 pair base
  const int d2b = t >> 4;                // 0..15
#pragma unroll
  for (int nn = 0; nn < N_ENS; ++nn) {
#pragma unroll
    for (int s = 0; s < 4; ++s) {
      const int dl = d2b + 16 * s;
      unsigned short lo = __builtin_bit_cast(unsigned short, wtile[nn][p2    ][dl]);
      unsigned short hi = __builtin_bit_cast(unsigned short, wtile[nn][p2 + 1][dl]);
      unsigned int pk = (unsigned int)lo | ((unsigned int)hi << 16);
      *(unsigned int*)(wT + (size_t)nn * DD + (size_t)(j0 + dl) * D1C + (i0 + p2)) = pk;
    }
  }
}

// ---------------------------------------------------------------------------
// Kernel 2: batched GEMM  act[b,n,l] = sum_d x[b,n,d] * w[n,d,l] + bias[n,l]
// 128x128 tile / block (8 wave32), K-step 32, double-buffered LDS, f16 WMMA.
// B operand copied global->LDS via async-to-LDS DMA when available (ASYNCcnt).
// ---------------------------------------------------------------------------
static __device__ inline v8h cvt8(v4f a, v4f b) {
  v8h r;
  r[0]=(h16)a[0]; r[1]=(h16)a[1]; r[2]=(h16)a[2]; r[3]=(h16)a[3];
  r[4]=(h16)b[0]; r[5]=(h16)b[1]; r[6]=(h16)b[2]; r[7]=(h16)b[3];
  return r;
}

__global__ __launch_bounds__(256) void baens_gemm_kernel(
    const float* __restrict__ x, const h16* __restrict__ wT,
    const float* __restrict__ bias, float* __restrict__ out)
{
  __shared__ h16 As[2][128][40];   // [buf][m][k], 8-half row pad
  __shared__ h16 Bs[2][128][40];   // [buf][ncol][k], 8-half row pad

  const int t  = threadIdx.x;
  const int n  = blockIdx.y;
  const int m0 = (blockIdx.x & 31) << 7;   // row-tile base (batch dim)
  const int c0 = (blockIdx.x >> 5) << 7;   // col-tile base (D2 dim)

  const int lane = t & 31;
  const int wave = t >> 5;
  const int lm   = lane & 15;
  const int lh   = lane >> 4;
  const int wmB  = (wave & 1) << 6;        // wave row base (0/64)
  const int wnB  = (wave >> 1) << 5;       // wave col base (0/32/64/96)

  const int ld_row = t >> 1;               // 0..127: A row / B col
  const int ld_seg = (t & 1) << 4;         // 0 or 16 (k offset)

  const float* xp = x  + (size_t)(m0 + ld_row) * (N_ENS * D1C) + n * D1C + ld_seg;
  const h16*   wp = wT + (size_t)n * DD + (size_t)(c0 + ld_row) * D1C + ld_seg;

  v8f acc[4][2];
#pragma unroll
  for (int mf = 0; mf < 4; ++mf)
#pragma unroll
    for (int nf = 0; nf < 2; ++nf)
#pragma unroll
      for (int e = 0; e < 8; ++e) acc[mf][nf][e] = 0.f;

  // prime buffer 0
  {
    v4f a0 = *(const v4f*)(xp + 0);
    v4f a1 = *(const v4f*)(xp + 4);
    v4f a2 = *(const v4f*)(xp + 8);
    v4f a3 = *(const v4f*)(xp + 12);
#if USE_ASYNC_B
    // async DMA: global -> LDS directly; imm offset applies to BOTH addresses
    __builtin_amdgcn_global_load_async_to_lds_b128(
        (AS1 v4i*)wp, (AS3 v4i*)&Bs[0][ld_row][ld_seg], 0, 0);
    __builtin_amdgcn_global_load_async_to_lds_b128(
        (AS1 v4i*)wp, (AS3 v4i*)&Bs[0][ld_row][ld_seg], 16, 0);
#else
    v8h b0 = *(const v8h*)(wp + 0);
    v8h b1 = *(const v8h*)(wp + 8);
    *(v8h*)&Bs[0][ld_row][ld_seg]     = b0;
    *(v8h*)&Bs[0][ld_row][ld_seg + 8] = b1;
#endif
    *(v8h*)&As[0][ld_row][ld_seg]     = cvt8(a0, a1);
    *(v8h*)&As[0][ld_row][ld_seg + 8] = cvt8(a2, a3);
  }
#if USE_ASYNC_B
  __builtin_amdgcn_s_wait_asynccnt(0);
#endif
  __syncthreads();

  for (int kt = 0; kt < 32; ++kt) {
    const int cur = kt & 1;
    const int nxt = cur ^ 1;
    // prefetch next K-tile (clamped on last iter; redundant but keeps EXEC uniform)
    const int ktn = (kt < 31) ? (kt + 1) : 31;
    const float* xq = xp + ktn * 32;
    v4f na0 = *(const v4f*)(xq + 0);
    v4f na1 = *(const v4f*)(xq + 4);
    v4f na2 = *(const v4f*)(xq + 8);
    v4f na3 = *(const v4f*)(xq + 12);
#if USE_ASYNC_B
    {
      const h16* wq = wp + ktn * 32;
      __builtin_amdgcn_global_load_async_to_lds_b128(
          (AS1 v4i*)wq, (AS3 v4i*)&Bs[nxt][ld_row][ld_seg], 0, 0);
      __builtin_amdgcn_global_load_async_to_lds_b128(
          (AS1 v4i*)wq, (AS3 v4i*)&Bs[nxt][ld_row][ld_seg], 16, 0);
    }
#else
    const h16* wq = wp + ktn * 32;
    v8h nb0 = *(const v8h*)(wq + 0);
    v8h nb1 = *(const v8h*)(wq + 8);
#endif

    // A fragments: elements 0..7 <- K = 8*lh + 0..7 ; 8..15 <- K = 16 + 8*lh + 0..7
    v16h afrag[4];
#pragma unroll
    for (int mf = 0; mf < 4; ++mf) {
      const int row = wmB + mf * 16 + lm;
      v8h lo = *(const v8h*)&As[cur][row][lh * 8];
      v8h hi = *(const v8h*)&As[cur][row][16 + lh * 8];
      afrag[mf] = __builtin_shufflevector(lo, hi, 0,1,2,3,4,5,6,7,8,9,10,11,12,13,14,15);
    }
    // B fragments: lane = column, elements 0..7 <- K = 8*lh + 0..7 ; 8..15 <- K = 16 + 8*lh + 0..7
    v16h bfrag[2];
#pragma unroll
    for (int nf = 0; nf < 2; ++nf) {
      const int col = wnB + nf * 16 + lm;
      v8h lo = *(const v8h*)&Bs[cur][col][lh * 8];
      v8h hi = *(const v8h*)&Bs[cur][col][16 + lh * 8];
      bfrag[nf] = __builtin_shufflevector(lo, hi, 0,1,2,3,4,5,6,7,8,9,10,11,12,13,14,15);
    }

#pragma unroll
    for (int mf = 0; mf < 4; ++mf)
#pragma unroll
      for (int nf = 0; nf < 2; ++nf)
        acc[mf][nf] = __builtin_amdgcn_wmma_f32_16x16x32_f16(
            false, afrag[mf], false, bfrag[nf],
            (short)0, acc[mf][nf], false, false);

    // stage next A tile into the other buffer
    *(v8h*)&As[nxt][ld_row][ld_seg]     = cvt8(na0, na1);
    *(v8h*)&As[nxt][ld_row][ld_seg + 8] = cvt8(na2, na3);
#if !USE_ASYNC_B
    *(v8h*)&Bs[nxt][ld_row][ld_seg]     = nb0;
    *(v8h*)&Bs[nxt][ld_row][ld_seg + 8] = nb1;
#endif
#if USE_ASYNC_B
    __builtin_amdgcn_s_wait_asynccnt(0);
#endif
    __syncthreads();
  }

  // epilogue: C layout -> VGPR r holds (M = 8*lh + r, N = lm) of each 16x16 tile
#pragma unroll
  for (int nf = 0; nf < 2; ++nf) {
    const int col = c0 + wnB + nf * 16 + lm;
    const float bv = bias[n * D2C + col];
#pragma unroll
    for (int mf = 0; mf < 4; ++mf) {
#pragma unroll
      for (int r = 0; r < 8; ++r) {
        const int row = m0 + wmB + mf * 16 + lh * 8 + r;
        out[(size_t)row * (N_ENS * D2C) + n * D2C + col] = acc[mf][nf][r] + bv;
      }
    }
  }
}

// ---------------------------------------------------------------------------
extern "C" void kernel_launch(void* const* d_in, const int* in_sizes, int n_in,
                              void* d_out, int out_size, void* d_ws, size_t ws_size,
                              hipStream_t stream) {
  (void)in_sizes; (void)n_in; (void)out_size; (void)ws_size;
  const float* x     = (const float*)d_in[0];  // (4096, 8, 1024)
  const float* U     = (const float*)d_in[1];  // (8, 1048576, 4)
  const float* V     = (const float*)d_in[2];  // (4, 1048576, 4)
  const float* scale = (const float*)d_in[3];  // (8, 1)
  const float* biasq = (const float*)d_in[4];  // (8, 1)
  const float* bias  = (const float*)d_in[5];  // (8, 1024)
  float* out = (float*)d_out;                  // (4096, 8, 1024)

  h16* wT = (h16*)d_ws;                        // 8*1024*1024 halves = 16 MB

  decode_w_kernel<<<dim3(16, 32), 256, 0, stream>>>(U, V, scale, biasq, wT);
  baens_gemm_kernel<<<dim3(256, 8), 256, 0, stream>>>(x, wT, bias, out);
}
